// Quantizer_88983132438921
// MI455X (gfx1250) — matrix operand order repaired and checked by reading, not compiled, not run on hardware
//
#include <hip/hip_runtime.h>

typedef __attribute__((ext_vector_type(16))) _Float16 v16h;
typedef __attribute__((ext_vector_type(8)))  float    v8f;

#define E_DIM    64
#define N_EMBED  8192
#define HW       1024            // 32*32
#define CHUNKS   (N_EMBED / 16)  // 512 column-chunks of 16
#define WAVES_PER_TILE 4         // column-split waves per block
#define CHUNKS_PER_WAVE (CHUNKS / WAVES_PER_TILE)  // 128
#define ROWS_PER_WAVE 32         // two 16-row M tiles per wave (2x B reuse)

// ---------------------------------------------------------------------------
// Prep 1: split embed[k][j] into f16 hi/lo and scatter into the contiguous-k WMMA
// B-fragment layout so the main loop does one 32B vector load per fragment.
// Fragment addressing for column-chunk c, K-slab s (K in [s*32, s*32+32)):
//   lane = (j%16) + 16*((k%32) >= 16);  half-slot = (k%32) & 15
//   halfs[ ((c*2 + s)*32 + lane)*16 + slot ]
// ---------------------------------------------------------------------------
__global__ void pack_embed_kernel(const float* __restrict__ embed,
                                  _Float16* __restrict__ bhi,
                                  _Float16* __restrict__ blo) {
  int id = blockIdx.x * 256 + threadIdx.x;     // 0 .. 64*8192-1
  int j = id & (N_EMBED - 1);
  int k = id >> 13;
  float v = embed[id];
  _Float16 hi = (_Float16)v;
  _Float16 lo = (_Float16)(v - (float)hi);
  int chunk = j >> 4;
  int s     = k >> 5;
  int kr    = k & 31;
  int lane  = (j & 15) + ((kr >= 16) ? 16 : 0);
  int slot  = kr & 15;
  int addr  = (((chunk * 2 + s) * 32) + lane) * 16 + slot;
  bhi[addr] = hi;
  blo[addr] = lo;
}

// Prep 2: bias[j] = 0.5 * ||e_j||^2  (score = f.e - bias; argmax == argmin dist)
__global__ void col_norms_kernel(const float* __restrict__ embed,
                                 float* __restrict__ norms) {
  int j = blockIdx.x * 256 + threadIdx.x;
  float s = 0.f;
#pragma unroll 8
  for (int k = 0; k < E_DIM; ++k) {
    float e = embed[(size_t)k * N_EMBED + j];
    s += e * e;
  }
  norms[j] = 0.5f * s;
}

// ---------------------------------------------------------------------------
// Main kernel: one block (4 waves) per 32-row tile; each wave owns a 2048-
// column slice and computes TWO 16x16 output tiles per chunk so every loaded
// B fragment is used twice (halves L2 fragment traffic). Split-f16 WMMA
// (3 products per K=32 slab) gives ~f32 accuracy on the f16 matrix pipe.
// ---------------------------------------------------------------------------
__global__ __launch_bounds__(128) void vq_argmin_gather_kernel(
    const float* __restrict__ x,          // [16, 64, 1024] NCHW flattened
    const float* __restrict__ embed,      // [64, 8192]
    const _Float16* __restrict__ bhi,
    const _Float16* __restrict__ blo,
    const float* __restrict__ norms,
    float* __restrict__ out) {            // [16384, 64] NHWC flattened
  __shared__ float s_val[WAVES_PER_TILE][ROWS_PER_WAVE];
  __shared__ int   s_idx[WAVES_PER_TILE][ROWS_PER_WAVE];
  __shared__ int   s_final[ROWS_PER_WAVE];

  const int tid  = threadIdx.x;
  const int wave = tid >> 5;
  const int lane = tid & 31;
  const int tile = blockIdx.x;            // 0..511
  const int n0   = tile * ROWS_PER_WAVE;  // first flat row of tile
  const int b    = n0 >> 10;              // batch (32 rows never straddle HW)
  const int hw0  = n0 & (HW - 1);

  // ---- Build A fragments (hi/lo split) for both 16-row sets, once per wave.
  // 16-bit A 16x32 layout (contiguous, matching packed B): lanes 0-15 row M=lane hold
  // K = h; lanes 16-31 get the +16 K offset.
  const int m     = lane & 15;
  const int kbase = (lane >= 16) ? 16 : 0;
  const float* xr0 = x + (size_t)b * E_DIM * HW + (hw0 + m);        // rows n0+m
  const float* xr1 = xr0 + 16;                                      // rows n0+16+m
  v16h ah00, ah01, al00, al01;   // row-set 0, K slabs 0/1
  v16h ah10, ah11, al10, al11;   // row-set 1, K slabs 0/1
#pragma unroll
  for (int h = 0; h < 16; ++h) {
    int k0 = kbase + h;
    float a, c;
    _Float16 fh;
    a = xr0[(size_t)k0 * HW];        fh = (_Float16)a; ah00[h] = fh; al00[h] = (_Float16)(a - (float)fh);
    a = xr0[(size_t)(k0 + 32) * HW]; fh = (_Float16)a; ah01[h] = fh; al01[h] = (_Float16)(a - (float)fh);
    c = xr1[(size_t)k0 * HW];        fh = (_Float16)c; ah10[h] = fh; al10[h] = (_Float16)(c - (float)fh);
    c = xr1[(size_t)(k0 + 32) * HW]; fh = (_Float16)c; ah11[h] = fh; al11[h] = (_Float16)(c - (float)fh);
  }

  float best0[8], best1[8];
  int   bidx0[8], bidx1[8];
#pragma unroll
  for (int r = 0; r < 8; ++r) {
    best0[r] = -3.0e38f; bidx0[r] = 0;
    best1[r] = -3.0e38f; bidx1[r] = 0;
  }

  const int cbeg = wave * CHUNKS_PER_WAVE;
  for (int c = cbeg; c < cbeg + CHUNKS_PER_WAVE; ++c) {
    const _Float16* bhp = bhi + (size_t)c * 1024 + lane * 16;  // 2 frags * 512 halfs
    const _Float16* blp = blo + (size_t)c * 1024 + lane * 16;
    v16h bh0 = *(const v16h*)bhp;
    v16h bh1 = *(const v16h*)(bhp + 512);
    v16h bl0 = *(const v16h*)blp;
    v16h bl1 = *(const v16h*)(blp + 512);
    __builtin_prefetch(bhp + 1024, 0, 3);   // next chunk (stays within ws)
    __builtin_prefetch(blp + 1024, 0, 3);

    // Four independent 3-deep accumulator chains: (row-set t) x (K slab s).
    v8f a00, a01, a10, a11;
    a00 = __builtin_amdgcn_wmma_f32_16x16x32_f16(false, ah00, false, bh0, (short)0, (v8f){}, false, false);
    a01 = __builtin_amdgcn_wmma_f32_16x16x32_f16(false, ah01, false, bh1, (short)0, (v8f){}, false, false);
    a10 = __builtin_amdgcn_wmma_f32_16x16x32_f16(false, ah10, false, bh0, (short)0, (v8f){}, false, false);
    a11 = __builtin_amdgcn_wmma_f32_16x16x32_f16(false, ah11, false, bh1, (short)0, (v8f){}, false, false);
    a00 = __builtin_amdgcn_wmma_f32_16x16x32_f16(false, ah00, false, bl0, (short)0, a00, false, false);
    a01 = __builtin_amdgcn_wmma_f32_16x16x32_f16(false, ah01, false, bl1, (short)0, a01, false, false);
    a10 = __builtin_amdgcn_wmma_f32_16x16x32_f16(false, ah10, false, bl0, (short)0, a10, false, false);
    a11 = __builtin_amdgcn_wmma_f32_16x16x32_f16(false, ah11, false, bl1, (short)0, a11, false, false);
    a00 = __builtin_amdgcn_wmma_f32_16x16x32_f16(false, al00, false, bh0, (short)0, a00, false, false);
    a01 = __builtin_amdgcn_wmma_f32_16x16x32_f16(false, al01, false, bh1, (short)0, a01, false, false);
    a10 = __builtin_amdgcn_wmma_f32_16x16x32_f16(false, al10, false, bh0, (short)0, a10, false, false);
    a11 = __builtin_amdgcn_wmma_f32_16x16x32_f16(false, al11, false, bh1, (short)0, a11, false, false);

    // C/D layout: lane holds column j = c*16 + (lane&15); VGPR r = row r + 8*(lane>>4)
    const int   j   = (c << 4) + (lane & 15);
    const float nrm = norms[j];
#pragma unroll
    for (int r = 0; r < 8; ++r) {
      float sc0 = a00[r] + a01[r] - nrm;
      float sc1 = a10[r] + a11[r] - nrm;
      if (sc0 > best0[r]) { best0[r] = sc0; bidx0[r] = j; }  // strict >: first occurrence
      if (sc1 > best1[r]) { best1[r] = sc1; bidx1[r] = j; }
    }
  }

  // Reduce over the 16 lanes of each half-wave (tie -> lower code index).
#pragma unroll
  for (int mask = 1; mask <= 8; mask <<= 1) {
#pragma unroll
    for (int r = 0; r < 8; ++r) {
      float ov; int oi;
      ov = __shfl_xor(best0[r], mask, 32);
      oi = __shfl_xor(bidx0[r], mask, 32);
      if (ov > best0[r] || (ov == best0[r] && oi < bidx0[r])) { best0[r] = ov; bidx0[r] = oi; }
      ov = __shfl_xor(best1[r], mask, 32);
      oi = __shfl_xor(bidx1[r], mask, 32);
      if (ov > best1[r] || (ov == best1[r] && oi < bidx1[r])) { best1[r] = ov; bidx1[r] = oi; }
    }
  }
  if ((lane & 15) == 0) {
    int rowoff = (lane >> 4) * 8;   // lane 0 -> rows 0..7, lane 16 -> rows 8..15
#pragma unroll
    for (int r = 0; r < 8; ++r) {
      s_val[wave][rowoff + r]      = best0[r];
      s_idx[wave][rowoff + r]      = bidx0[r];
      s_val[wave][16 + rowoff + r] = best1[r];
      s_idx[wave][16 + rowoff + r] = bidx1[r];
    }
  }
  __syncthreads();

  if (tid < ROWS_PER_WAVE) {  // combine waves (ascending j ranges; strict > keeps lowest j)
    float bv = s_val[0][tid];
    int   bi = s_idx[0][tid];
#pragma unroll
    for (int w = 1; w < WAVES_PER_TILE; ++w) {
      float v = s_val[w][tid];
      int   i = s_idx[w][tid];
      if (v > bv) { bv = v; bi = i; }
    }
    s_final[tid] = bi;
  }
  __syncthreads();

  // Gather winning codes: out[(n0+mm)*64 + e] = embed[e*8192 + idx]; coalesced stores.
#pragma unroll
  for (int i = 0; i < 16; ++i) {
    int el = tid + i * 128;   // 0..2047 over 32 rows x 64 channels
    int mm = el >> 6;
    int e  = el & 63;
    int idx = s_final[mm];
    out[(size_t)(n0 + mm) * E_DIM + e] = embed[(size_t)e * N_EMBED + idx];
  }
}

// ---------------------------------------------------------------------------
extern "C" void kernel_launch(void* const* d_in, const int* in_sizes, int n_in,
                              void* d_out, int out_size, void* d_ws, size_t ws_size,
                              hipStream_t stream) {
  const float* x     = (const float*)d_in[0];   // [16,64,32,32]
  const float* embed = (const float*)d_in[1];   // [64,8192]
  float* out = (float*)d_out;                   // [16,32,32,64]

  char* ws = (char*)d_ws;
  _Float16* bhi  = (_Float16*)(ws);                        // 1 MiB
  _Float16* blo  = (_Float16*)(ws + (1u << 20));           // 1 MiB
  float*    nrms = (float*)   (ws + (2u << 20));           // 32 KiB

  pack_embed_kernel<<<(E_DIM * N_EMBED) / 256, 256, 0, stream>>>(embed, bhi, blo);
  col_norms_kernel <<<N_EMBED / 256,           256, 0, stream>>>(embed, nrms);
  vq_argmin_gather_kernel<<<16384 / ROWS_PER_WAVE, 128, 0, stream>>>(x, embed, bhi, blo, nrms, out);
}
